// SelfAttention_28552942584285
// MI455X (gfx1250) — matrix operand order repaired and checked by reading, not compiled, not run on hardware
//
#include <hip/hip_runtime.h>
#include <hip/hip_bf16.h>
#include <cstddef>

// B=32, S=1024, D=256 self-attention:
//   s = x @ x^T ; scores = exp(s)/16 ; p = softmax(scores) ; out = (p @ x).sum(rows)
// d_out = [ out (32*256 f32) | p_attn (32*1024*1024 f32) ]
// Bandwidth-bound on the 128MB p_attn store -> f16 WMMA (f32 acc) is ample.

typedef __attribute__((ext_vector_type(16))) _Float16 v16h;
typedef __attribute__((ext_vector_type(8)))  float    v8f;

union AFrag { v16h v; _Float16 e[16]; };

__global__ __launch_bounds__(256)
void attn_kernel(const float* __restrict__ x, float* __restrict__ pbuf,
                 float* __restrict__ ws)
{
    constexpr int S = 1024, D = 256, BM = 64;

    __shared__ _Float16 qs[BM * D];    // 32 KB : Q tile (rows of this block)
    __shared__ _Float16 kb[32 * D];    // 16 KB : streamed K / X tile (32 t-rows)
    __shared__ _Float16 ps[BM * 32];   //  4 KB : p chunk (f16) for PV WMMA
    __shared__ float rmaxp[2 * BM];
    __shared__ float vmax[BM];
    __shared__ float rinv[BM];
    __shared__ float rsum4[BM * 4];
    __shared__ float colacc[4 * D];    //  4 KB

    const int tid   = threadIdx.x;
    const int lane  = tid & 31;        // wave32 on gfx1250
    const int wid   = tid >> 5;        // 8 waves
    const int b     = blockIdx.y;
    const int rtile = blockIdx.x;
    const int row0  = rtile * BM;

    const int m  = lane & 15;          // A: row-in-strip / B: N / D: col
    const int hi = lane >> 4;          // lane half selects K/row halves

    const int strip = wid >> 1;        // 4 row strips of 16
    const int ch    = wid & 1;         // col half (pass A: 16 t-cols; pass C: 128 D-cols)

    // ---- stage Q tile: x[b, row0:row0+64, :] f32 -> f16 LDS ----
    {
        const float4* src = reinterpret_cast<const float4*>(x + ((size_t)b * S + row0) * D);
        #pragma unroll
        for (int j = 0; j < 16; ++j) {
            int i4 = tid + j * 256;                 // 4096 float4
            float4 v = src[i4];
            kbNOP:;
            qs[i4 * 4 + 0] = (_Float16)v.x;
            qs[i4 * 4 + 1] = (_Float16)v.y;
            qs[i4 * 4 + 2] = (_Float16)v.z;
            qs[i4 * 4 + 3] = (_Float16)v.w;
        }
    }
    __syncthreads();

    // =========== Pass A: scores s = q . k, write raw s, track row max ==========
    float rmax[8];
    #pragma unroll
    for (int r = 0; r < 8; ++r) rmax[r] = -3.0e38f;

    for (int tc = 0; tc < 32; ++tc) {
        { // stage K tile: x[b, tc*32 : +32, :]
            const float4* src = reinterpret_cast<const float4*>(x + ((size_t)b * S + tc * 32) * D);
            #pragma unroll
            for (int j = 0; j < 8; ++j) {
                int i4 = tid + j * 256;             // 2048 float4
                float4 v = src[i4];
                kb[i4 * 4 + 0] = (_Float16)v.x;
                kb[i4 * 4 + 1] = (_Float16)v.y;
                kb[i4 * 4 + 2] = (_Float16)v.z;
                kb[i4 * 4 + 3] = (_Float16)v.w;
            }
            if (tc + 1 < 32)  // global_prefetch_b8 for the next tile
                __builtin_prefetch(x + ((size_t)b * S + (tc + 1) * 32) * D + tid * 8, 0, 1);
        }
        __syncthreads();

        v8f acc = {};
        #pragma unroll
        for (int kc = 0; kc < 8; ++kc) {            // contraction over D=256 in K=32 steps
            const int K0 = kc * 32;
            AFrag a, bb;
            #pragma unroll
            for (int i = 0; i < 16; ++i) {          // A layout used here (contiguous): K = hi*16 + i
                int ko = hi * 16 + i;
                a.e[i] = qs[(strip * 16 + m) * D + K0 + ko];
            }
            #pragma unroll
            for (int i = 0; i < 16; ++i)            // B layout used here (same as A): K = hi*16 + i, N = m
                bb.e[i] = kb[(ch * 16 + m) * D + K0 + hi * 16 + i];
            acc = __builtin_amdgcn_wmma_f32_16x16x32_f16(
                      false, a.v, false, bb.v, (short)0, acc, false, false);
        }
        { // D layout: row = hi*8 + r, col = m  -> write s to p_attn scratch
            const int gcol = tc * 32 + ch * 16 + m;
            #pragma unroll
            for (int r = 0; r < 8; ++r) {
                int grow = row0 + strip * 16 + hi * 8 + r;
                pbuf[((size_t)b * S + grow) * S + gcol] = acc[r];
                rmax[r] = fmaxf(rmax[r], acc[r]);
            }
        }
        __syncthreads();
    }

    // row max: reduce across 16-lane group (cols), publish per col-half, combine
    #pragma unroll
    for (int r = 0; r < 8; ++r) {
        float v = rmax[r];
        v = fmaxf(v, __shfl_xor(v, 1, 32));
        v = fmaxf(v, __shfl_xor(v, 2, 32));
        v = fmaxf(v, __shfl_xor(v, 4, 32));
        v = fmaxf(v, __shfl_xor(v, 8, 32));
        rmax[r] = v;
    }
    if (m == 0) {
        #pragma unroll
        for (int r = 0; r < 8; ++r)
            rmaxp[ch * BM + strip * 16 + hi * 8 + r] = rmax[r];
    }
    __threadfence();   // make pass-A s stores visible before re-reading
    __syncthreads();
    if (tid < BM)      // softmax operates on v = exp(s)/16 ; max is monotone in s
        vmax[tid] = __expf(fmaxf(rmaxp[tid], rmaxp[BM + tid])) * 0.0625f;
    __syncthreads();

    // =========== Pass B: row sums of exp(v - vmax) ==========
    {
        const int row  = tid >> 2;
        const int part = tid & 3;
        const float vm = vmax[row];
        const float* srow = pbuf + ((size_t)b * S + row0 + row) * S + part * 256;
        float sum = 0.f;
        for (int c = 0; c < 256; ++c)
            sum += __expf(__expf(srow[c]) * 0.0625f - vm);
        rsum4[row * 4 + part] = sum;
    }
    __syncthreads();
    if (tid < BM)
        rinv[tid] = 1.0f / (rsum4[tid * 4] + rsum4[tid * 4 + 1] +
                            rsum4[tid * 4 + 2] + rsum4[tid * 4 + 3]);
    __syncthreads();

    // =========== Pass C: p = e/rowsum (store), out += p @ x via WMMA ==========
    v8f oacc[8];
    #pragma unroll
    for (int nt = 0; nt < 8; ++nt) oacc[nt] = {};

    for (int tc = 0; tc < 32; ++tc) {
        // finalize p for this 64x32 chunk: overwrite p_attn, stash f16 copy
        #pragma unroll
        for (int j = 0; j < 8; ++j) {
            int e = tid + j * 256;
            int row = e >> 5, col = e & 31;
            size_t gi = ((size_t)b * S + row0 + row) * S + tc * 32 + col;
            float p = __expf(__expf(pbuf[gi]) * 0.0625f - vmax[row]) * rinv[row];
            pbuf[gi] = p;
            ps[row * 32 + col] = (_Float16)p;
        }
        { // stage x tile rows [tc*32, +32)
            const float4* src = reinterpret_cast<const float4*>(x + ((size_t)b * S + tc * 32) * D);
            #pragma unroll
            for (int j = 0; j < 8; ++j) {
                int i4 = tid + j * 256;
                float4 v = src[i4];
                kb[i4 * 4 + 0] = (_Float16)v.x;
                kb[i4 * 4 + 1] = (_Float16)v.y;
                kb[i4 * 4 + 2] = (_Float16)v.z;
                kb[i4 * 4 + 3] = (_Float16)v.w;
            }
        }
        __syncthreads();

        AFrag a;
        #pragma unroll
        for (int i = 0; i < 16; ++i) {              // A = p chunk, K = t-in-chunk
            int ko = hi * 16 + i;
            a.e[i] = ps[(strip * 16 + m) * 32 + ko];
        }
        #pragma unroll
        for (int nt = 0; nt < 8; ++nt) {            // wave owns D cols [ch*128, +128)
            AFrag bb;
            #pragma unroll
            for (int i = 0; i < 16; ++i)            // B = x chunk, K = t, N = d
                bb.e[i] = kb[(hi * 16 + i) * D + ch * 128 + nt * 16 + m];
            oacc[nt] = __builtin_amdgcn_wmma_f32_16x16x32_f16(
                           false, a.v, false, bb.v, (short)0, oacc[nt], false, false);
        }
        __syncthreads();
    }

    // sum over the 16 output rows per wave, then over 4 strips via LDS
    #pragma unroll
    for (int nt = 0; nt < 8; ++nt) {
        float s = 0.f;
        #pragma unroll
        for (int r = 0; r < 8; ++r) s += oacc[nt][r];
        s += __shfl_xor(s, 16, 32);                 // combine row halves (same N)
        if (hi == 0)
            colacc[strip * D + ch * 128 + nt * 16 + m] = s;
    }
    __syncthreads();
    if (tid < D) {
        float s = colacc[tid] + colacc[D + tid] + colacc[2 * D + tid] + colacc[3 * D + tid];
        ws[((size_t)b * 16 + rtile) * D + tid] = s;
    }
}

// out[b,d] = sum over the 16 row-tiles' partial column sums (deterministic)
__global__ __launch_bounds__(256)
void attn_reduce(const float* __restrict__ ws, float* __restrict__ out)
{
    int idx = blockIdx.x * 256 + threadIdx.x;       // 8192 = 32*256
    int b = idx >> 8, d = idx & 255;
    float s = 0.f;
    #pragma unroll
    for (int rt = 0; rt < 16; ++rt)
        s += ws[((size_t)b * 16 + rt) * 256 + d];
    out[idx] = s;
}

extern "C" void kernel_launch(void* const* d_in, const int* in_sizes, int n_in,
                              void* d_out, int out_size, void* d_ws, size_t ws_size,
                              hipStream_t stream)
{
    (void)in_sizes; (void)n_in; (void)out_size; (void)ws_size;
    const float* x    = (const float*)d_in[0];
    float*       out  = (float*)d_out;              // [32*256] then p_attn
    float*       pbuf = out + 32 * 256;             // [32,1024,1024]
    float*       ws   = (float*)d_ws;               // 512*256 f32 partials

    dim3 grid(16, 32);                              // (row tiles, batch)
    attn_kernel<<<grid, 256, 0, stream>>>(x, pbuf, ws);
    attn_reduce<<<32, 256, 0, stream>>>(ws, out);
}
